// MyModel_87522843560808
// MI455X (gfx1250) — compile-verified, matrix operand order checked by reading, not run on hardware
//
#include <hip/hip_runtime.h>
#include <hip/hip_bf16.h>

typedef __attribute__((ext_vector_type(16))) _Float16 v16h;
typedef __attribute__((ext_vector_type(8)))  _Float16 v8h;
typedef __attribute__((ext_vector_type(8)))  float    v8f;

#define B_  256
#define T_  512
#define E_  128
#define U_  128
#define G4  512  // 4*U

// ---------------------------------------------------------------------------
// tanh via exp, safe at +/-inf: tanh(x) = 1 - 2/(exp(2x)+1)
__device__ __forceinline__ float tanh_fast(float x) {
    return 1.0f - 2.0f / (__expf(2.0f * x) + 1.0f);
}
__device__ __forceinline__ float sigmoid_fast(float x) {
    return 1.0f / (1.0f + __expf(-x));
}

// ---------------------------------------------------------------------------
// Kernel 0: pack f32 [128,512] row-major weight into f16 WMMA B-fragment order.
// Layout: P[((kt*32+nt)*32 + lane)*16 + e], where for a 16-bit 32x16 B tile:
//   n = nt*16 + (lane&15), hi = lane>>4,
//   k = kt*32 + (e<8 ? hi*8+e : 16 + hi*8 + (e-8))
__global__ void pack_weights(const float* __restrict__ Wk,
                             const float* __restrict__ Wr,
                             _Float16* __restrict__ Pk,
                             _Float16* __restrict__ Pr) {
    int id  = blockIdx.x * blockDim.x + threadIdx.x;   // 0 .. 2*65536-1
    int mat = id >> 16;
    int r   = id & 0xFFFF;
    int e    = r & 15;
    int lane = (r >> 4) & 31;
    int nt   = (r >> 9) & 31;
    int kt   = (r >> 14) & 3;
    int hi   = lane >> 4;
    int k = kt * 32 + ((e < 8) ? (hi * 8 + e) : (16 + hi * 8 + (e - 8)));
    int n = nt * 16 + (lane & 15);
    const float* W = mat ? Wr : Wk;
    _Float16*    P = mat ? Pr : Pk;
    P[r] = (_Float16)W[k * G4 + n];
}

// ---------------------------------------------------------------------------
// Kernel 1: embedding gather + zx = emb[tok] @ kernel + bias.
// Output is stored as f16 in WMMA-accumulator *fragment order*:
//   zxf[ ((tile*32 + nt)*32 + lane)*8 + r ],  tile = t*16 + btile
// so both this store and the scan's load are single contiguous b128 per lane.
__global__ __launch_bounds__(256)
void zx_gemm(const int* __restrict__ tokens, const float* __restrict__ emb,
             const _Float16* __restrict__ Pk, const float* __restrict__ bias,
             _Float16* __restrict__ zxf) {
    const int lane  = threadIdx.x & 31;
    const int wave  = threadIdx.x >> 5;
    const int tile  = blockIdx.x;          // 0..8191 ; rid = tile*16 + row
    const int rid0  = tile * 16;           // rid = t*256 + b
    const int t     = rid0 >> 8;
    const int bbase = rid0 & 255;
    const int m     = lane & 15;
    const int hi    = lane >> 4;

    const int tok = tokens[(bbase + m) * T_ + t];
    const float* erow = emb + (size_t)tok * E_;

    // A fragments (16x32 f16) for all 4 k-tiles, converted f32->f16
    v16h a[4];
#pragma unroll
    for (int kt = 0; kt < 4; ++kt) {
        const float* p0 = erow + kt * 32 + hi * 8;        // e = 0..7
        const float* p1 = erow + kt * 32 + 16 + hi * 8;   // e = 8..15
#pragma unroll
        for (int e = 0; e < 8; ++e) a[kt][e]     = (_Float16)p0[e];
#pragma unroll
        for (int e = 0; e < 8; ++e) a[kt][8 + e] = (_Float16)p1[e];
    }

    const size_t tilebase = (size_t)tile * 32 * 32 * 8;   // halves
#pragma unroll
    for (int i = 0; i < 4; ++i) {
        const int nt = wave * 4 + i;
        v8f acc = {};
#pragma unroll
        for (int kt = 0; kt < 4; ++kt) {
            v16h bf = *(const v16h*)(Pk + ((kt * 32 + nt) * 32 + lane) * 16);
            acc = __builtin_amdgcn_wmma_f32_16x16x32_f16(
                false, a[kt], false, bf, (short)0, acc, false, false);
        }
        const float bv = bias[nt * 16 + m];
        v8h hv;
#pragma unroll
        for (int r = 0; r < 8; ++r) hv[r] = (_Float16)(acc[r] + bv);
        *(v8h*)(zxf + tilebase + ((size_t)nt * 32 + lane) * 8) = hv;
    }
}

// ---------------------------------------------------------------------------
// Kernel 2: persistent LSTM scan. 16 blocks x 256 threads; block owns 16 batch
// rows. Each wave's 16 recurrent B-fragments (its 4 gate tiles x 4 k-tiles)
// are held in VGPRs for all 512 steps (gfx1250 waves address up to 1024 VGPRs).
// Cell state c: one v8f per lane, in registers. h lives in 4 KB LDS as f16 so
// A-fragment rebuild is pure contiguous ds_load_b128 with no conversions.
// Wave w owns u-columns [16w,16w+16): N-tiles {w,8+w,16+w,24+w} are the
// i/f/g/o slices at identical lane positions -> lane-local gate math.
__global__ __launch_bounds__(256)
void lstm_scan(const _Float16* __restrict__ Pr, const _Float16* __restrict__ zxf,
               float* __restrict__ hfinal) {
    __shared__ __align__(16) _Float16 sh[16 * 128];   // 4 KB f16 h state

    const int lane = threadIdx.x & 31;
    const int wave = threadIdx.x >> 5;
    const int m    = lane & 15;
    const int hi   = lane >> 4;
    const int bt   = blockIdx.x;                 // batch tile 0..15
    const int u    = wave * 16 + m;              // this lane's hidden column

    // preload this wave's 16 recurrent B fragments into registers (128 VGPRs)
    v16h barr[4][4];
#pragma unroll
    for (int g = 0; g < 4; ++g) {
        const int nt = g * 8 + wave;
#pragma unroll
        for (int kt = 0; kt < 4; ++kt)
            barr[g][kt] = *(const v16h*)(Pr + ((kt * 32 + nt) * 32 + lane) * 16);
    }

    for (int i = threadIdx.x; i < 16 * 128; i += 256) sh[i] = (_Float16)0.0f;
    __syncthreads();

    v8f cc = {};   // cell state for rows (r+8*hi), column u

    for (int t = 0; t < T_; ++t) {
        // input projections for this step: one b128 per gate, fully coalesced
        const _Float16* zbase = zxf + (size_t)(t * 16 + bt) * 32 * 32 * 8;
        v8h zp[4];
#pragma unroll
        for (int g = 0; g < 4; ++g)
            zp[g] = *(const v8h*)(zbase + ((g * 8 + wave) * 32 + lane) * 8);

        if (t + 1 < T_) {   // pull next timestep's fragments toward the caches
            const _Float16* znext = zxf + (size_t)((t + 1) * 16 + bt) * 32 * 32 * 8;
#pragma unroll
            for (int g = 0; g < 4; ++g)
                __builtin_prefetch(znext + ((g * 8 + wave) * 32 + lane) * 8, 0, 0);
        }

        // A fragments straight from f16 h in LDS (2 x b128 per k-tile)
        v16h a[4];
#pragma unroll
        for (int kt = 0; kt < 4; ++kt) {
            v8h lo  = *(const v8h*)(&sh[m * 128 + kt * 32 + hi * 8]);
            v8h hi8 = *(const v8h*)(&sh[m * 128 + kt * 32 + 16 + hi * 8]);
#pragma unroll
            for (int e = 0; e < 8; ++e) { a[kt][e] = lo[e]; a[kt][8 + e] = hi8[e]; }
        }

        // z_gate = h @ R : kt-outer so consecutive WMMAs hit different accs
        v8f acc[4] = {};
#pragma unroll
        for (int kt = 0; kt < 4; ++kt)
#pragma unroll
            for (int g = 0; g < 4; ++g)
                acc[g] = __builtin_amdgcn_wmma_f32_16x16x32_f16(
                    false, a[kt], false, barr[g][kt], (short)0, acc[g], false, false);

        __syncthreads();   // everyone done reading old h
#pragma unroll
        for (int r = 0; r < 8; ++r) {
            const float ig = sigmoid_fast(acc[0][r] + (float)zp[0][r]);
            const float fg = sigmoid_fast(acc[1][r] + (float)zp[1][r]);
            const float gg = tanh_fast   (acc[2][r] + (float)zp[2][r]);
            const float og = sigmoid_fast(acc[3][r] + (float)zp[3][r]);
            const float cn = fg * cc[r] + ig * gg;
            cc[r] = cn;
            sh[(r + 8 * hi) * 128 + u] = (_Float16)(og * tanh_fast(cn));
        }
        __syncthreads();   // new h visible
    }

    for (int i = threadIdx.x; i < 16 * 128; i += 256)
        hfinal[(size_t)(bt * 16 + (i >> 7)) * U_ + (i & 127)] = (float)sh[i];
}

// ---------------------------------------------------------------------------
// Kernel 3: out[b] = sigmoid(h_final[b] . dense_w + dense_b)
__global__ void head(const float* __restrict__ hfinal, const float* __restrict__ w,
                     const float* __restrict__ bptr, float* __restrict__ out) {
    int b = blockIdx.x * blockDim.x + threadIdx.x;
    if (b >= B_) return;
    const float* hr = hfinal + (size_t)b * U_;
    float s = 0.0f;
#pragma unroll 8
    for (int uu = 0; uu < U_; ++uu) s += hr[uu] * w[uu];
    out[b] = sigmoid_fast(s + bptr[0]);
}

// ---------------------------------------------------------------------------
extern "C" void kernel_launch(void* const* d_in, const int* in_sizes, int n_in,
                              void* d_out, int out_size, void* d_ws, size_t ws_size,
                              hipStream_t stream) {
    const int*   tokens = (const int*)  d_in[0];
    const float* emb    = (const float*)d_in[1];
    const float* Wk     = (const float*)d_in[2];
    const float* Wr     = (const float*)d_in[3];
    const float* bias   = (const float*)d_in[4];
    const float* dw     = (const float*)d_in[5];
    const float* db     = (const float*)d_in[6];
    float* out = (float*)d_out;

    char* ws = (char*)d_ws;
    _Float16* Pk     = (_Float16*)(ws);              // 128 KB
    _Float16* Pr     = (_Float16*)(ws + 131072);     // 128 KB
    float*    hfinal = (float*)   (ws + 262144);     // 128 KB
    _Float16* zxf    = (_Float16*)(ws + 393216);     // 128 MB f16 fragment order

    pack_weights<<<512, 256, 0, stream>>>(Wk, Wr, Pk, Pr);
    zx_gemm   <<<8192, 256, 0, stream>>>(tokens, emb, Pk, bias, zxf);
    lstm_scan <<<16,   256, 0, stream>>>(Pr, zxf, hfinal);
    head      <<<1,    256, 0, stream>>>(hfinal, dw, db, out);
}
